// FloReLBlock_3204045603289
// MI455X (gfx1250) — matrix operand order repaired and checked by reading, not compiled, not run on hardware
//
#include <hip/hip_runtime.h>
#include <stdint.h>
#include <stddef.h>

// ---------------------------------------------------------------------------
// Types for CDNA5 WMMA (gfx1250, wave32)
// ---------------------------------------------------------------------------
typedef __bf16 bf16;
typedef __attribute__((ext_vector_type(4)))  __bf16 v4bf;
typedef __attribute__((ext_vector_type(8)))  __bf16 v8bf;
typedef __attribute__((ext_vector_type(16))) __bf16 v16bf;
typedef __attribute__((ext_vector_type(8)))  float  v8f;

static constexpr int BM  = 128;
static constexpr int BN  = 128;
static constexpr int BK  = 32;
static constexpr int LDT = 40;   // LDS row stride in bf16 elems (80B, keeps 16B alignment)

// ---------------------------------------------------------------------------
// Fused WMMA GEMM:  C[M,N] = epilogue( A[M,K] (f32) @ Bw (bf16, [n][k] row-major) )
//   MODE 0: C = tanh(acc + bias[n] + tscal*tvec[n])          (z@W1 stage -> h)
//   MODE 1: C = acc + bias[n]                                 (h@W2 + b2 -> f)
//   MODE 2: C = acc * (1 - h[m,n]^2)                          (e@W2^T -> g)
//   MODE 3: no C store; div[m] += sum_n acc[m,n]*e[m,n]       (g@W1^T -> trace)
// Assumes M%128==0, N%128==0, K%32==0 (true here: 2048 x {512,1024}).
// B tile is staged memory->LDS with GLOBAL_LOAD_ASYNC_TO_LDS_B128 (ASYNCcnt),
// overlapping with the A tile's f32->bf16 VGPR-path conversion.
// ---------------------------------------------------------------------------
template<int MODE>
__global__ __launch_bounds__(256)
void k_gemm(const float* __restrict__ A, const bf16* __restrict__ Bw,
            float* __restrict__ C, int M, int N, int K,
            const float* __restrict__ bias, const float* __restrict__ tvec, float tscal,
            const float* __restrict__ hbuf, const float* __restrict__ ebuf,
            float* __restrict__ divbuf)
{
  __shared__ bf16 sA[BM * LDT];
  __shared__ bf16 sB[BN * LDT];

  const int tid  = threadIdx.x;
  const int lane = tid & 31;
  const int wid  = tid >> 5;          // 8 waves (wave32)
  const int hi16 = lane >> 4;         // half-wave select
  const int l15  = lane & 15;
  const int mblk = blockIdx.y * BM;
  const int nblk = blockIdx.x * BN;
  const int wm   = (wid & 3) * 32;    // wave tile: 32 rows x 64 cols
  const int wn   = (wid >> 2) * 64;

  v8f acc[2][4];
#pragma unroll
  for (int mi = 0; mi < 2; ++mi)
#pragma unroll
    for (int ni = 0; ni < 4; ++ni)
#pragma unroll
      for (int k = 0; k < 8; ++k) acc[mi][ni][k] = 0.0f;

  float dotacc[2][8];
#pragma unroll
  for (int mi = 0; mi < 2; ++mi)
#pragma unroll
    for (int j = 0; j < 8; ++j) dotacc[mi][j] = 0.0f;

  for (int k0 = 0; k0 < K; k0 += BK) {
    // ---- B tile: async DMA memory -> LDS (no VGPR data path, ASYNCcnt)
#pragma unroll
    for (int it = 0; it < 2; ++it) {
      int bid = it * 2048 + tid * 8;
      int n = bid >> 5;
      int c = bid & 31;
      unsigned long long ga = (unsigned long long)(const void*)(
          Bw + (size_t)(nblk + n) * K + (size_t)(k0 + c));
      unsigned la = (unsigned)(size_t)(&sB[n * LDT + c]);
      asm volatile("global_load_async_to_lds_b128 %0, %1, off"
                   :: "v"(la), "v"(ga) : "memory");
    }
    // ---- A tile: f32 global -> cvt bf16 -> LDS (overlaps with async B DMA)
#pragma unroll
    for (int it = 0; it < 4; ++it) {
      int fid = it * 1024 + tid * 4;
      int r = fid >> 5;
      int c = fid & 31;
      const float4 v = *reinterpret_cast<const float4*>(
          A + (size_t)(mblk + r) * K + (size_t)(k0 + c));
      v4bf bv;
      bv[0] = (bf16)v.x; bv[1] = (bf16)v.y; bv[2] = (bf16)v.z; bv[3] = (bf16)v.w;
      *reinterpret_cast<v4bf*>(&sA[r * LDT + c]) = bv;
    }
    // ---- warm L2->WGP path for the next A tile while this K-step computes
    if (k0 + BK < K && tid < BM) {
      __builtin_prefetch(A + (size_t)(mblk + tid) * K + (size_t)(k0 + BK), 0, 3);
    }
    // our wave's async B transfers must be LDS-visible before the barrier
    asm volatile("s_wait_asynccnt 0x0" ::: "memory");
    __syncthreads();

    // ---- build WMMA fragments per ISA 16-bit layouts
    v16bf afr[2], bfr[4];
    const int kbA = hi16 * 8;    // A: lanes<16 hold K 0-7 & 16-23; lanes>=16 K 8-15 & 24-31
    const int kbB = hi16 * 8;    // B: lanes<16 hold K 0-7 & 16-23; lanes>=16 K 8-15 & 24-31
#pragma unroll
    for (int mi = 0; mi < 2; ++mi) {
      const bf16* p = &sA[(wm + mi * 16 + l15) * LDT];
      v8bf lo = *reinterpret_cast<const v8bf*>(p + kbA);
      v8bf hi = *reinterpret_cast<const v8bf*>(p + kbA + 16);
      afr[mi] = __builtin_shufflevector(lo, hi, 0,1,2,3,4,5,6,7,8,9,10,11,12,13,14,15);
    }
#pragma unroll
    for (int ni = 0; ni < 4; ++ni) {
      const bf16* p = &sB[(wn + ni * 16 + l15) * LDT];
      v8bf lo = *reinterpret_cast<const v8bf*>(p + kbB);
      v8bf hi = *reinterpret_cast<const v8bf*>(p + kbB + 16);
      bfr[ni] = __builtin_shufflevector(lo, hi, 0,1,2,3,4,5,6,7,8,9,10,11,12,13,14,15);
    }

    // ---- 8x v_wmma_f32_16x16x32_bf16 per K step
#pragma unroll
    for (int mi = 0; mi < 2; ++mi)
#pragma unroll
      for (int ni = 0; ni < 4; ++ni)
        acc[mi][ni] = __builtin_amdgcn_wmma_f32_16x16x32_bf16(
            false, afr[mi], false, bfr[ni], (short)0, acc[mi][ni], false, false);
    __syncthreads();
  }

  // ---- epilogue (C/D layout: lane l -> col l&15; VGPR j -> row (l<16? j : 8+j))
#pragma unroll
  for (int mi = 0; mi < 2; ++mi) {
#pragma unroll
    for (int ni = 0; ni < 4; ++ni) {
      const int col = nblk + wn + ni * 16 + l15;
#pragma unroll
      for (int j = 0; j < 8; ++j) {
        const int row = mblk + wm + mi * 16 + hi16 * 8 + j;
        float v = acc[mi][ni][j];
        if constexpr (MODE == 0) {
          C[(size_t)row * N + col] = tanhf(v + bias[col] + tscal * tvec[col]);
        } else if constexpr (MODE == 1) {
          C[(size_t)row * N + col] = v + bias[col];
        } else if constexpr (MODE == 2) {
          float hh = hbuf[(size_t)row * N + col];
          C[(size_t)row * N + col] = v * (1.0f - hh * hh);
        } else {
          dotacc[mi][j] += v * ebuf[(size_t)row * N + col];
        }
      }
    }
  }

  if constexpr (MODE == 3) {
    // reduce the e.(J^T e) partial dot across the 16 lanes holding one row
#pragma unroll
    for (int mi = 0; mi < 2; ++mi)
#pragma unroll
      for (int j = 0; j < 8; ++j) {
        float v = dotacc[mi][j];
        v += __shfl_xor(v, 1, 32);
        v += __shfl_xor(v, 2, 32);
        v += __shfl_xor(v, 4, 32);
        v += __shfl_xor(v, 8, 32);
        if (l15 == 0) {
          const int row = mblk + wm + mi * 16 + hi16 * 8 + j;
          atomicAdd(&divbuf[row], v);   // global_atomic_add_f32
        }
      }
  }
}

// ---------------------------------------------------------------------------
// Helper kernels
// ---------------------------------------------------------------------------
__global__ void k_cvt(const float* __restrict__ in, bf16* __restrict__ out, size_t n) {
  size_t i = (size_t)blockIdx.x * blockDim.x + threadIdx.x;
  if (i < n) out[i] = (bf16)in[i];
}

// in: R x C row-major f32; out: C x R row-major bf16 (transpose+convert)
__global__ void k_cvtT(const float* __restrict__ in, bf16* __restrict__ out, int R, int C) {
  size_t i = (size_t)blockIdx.x * blockDim.x + threadIdx.x;
  if (i >= (size_t)R * C) return;
  int r = (int)(i / C), c = (int)(i % C);
  out[(size_t)c * R + r] = (bf16)in[i];
}

__device__ __forceinline__ unsigned mix32(unsigned x) {
  x ^= x >> 16; x *= 0x7feb352dU; x ^= x >> 15; x *= 0x846ca68bU; x ^= x >> 16;
  return x;
}

__global__ void k_gen_e(float* __restrict__ e, size_t n, unsigned seed) {
  size_t i = (size_t)blockIdx.x * blockDim.x + threadIdx.x;
  if (i >= n) return;
  unsigned h = mix32((unsigned)i + seed * 0x9E3779B9u + 0x85EBCA6Bu);
  e[i] = (h & 0x10000u) ? 1.0f : -1.0f;
}

__global__ void k_zero(float* __restrict__ p, int n) {
  int i = blockIdx.x * blockDim.x + threadIdx.x;
  if (i < n) p[i] = 0.0f;
}

// out = ca*a [+ cb*b + cc*c + cd*d + ce*e], nt = number of terms (1..5)
__global__ void k_comb(float* __restrict__ out, size_t n,
                       const float* __restrict__ a, float ca,
                       const float* __restrict__ b, float cb,
                       const float* __restrict__ c, float cc,
                       const float* __restrict__ d, float cd_,
                       const float* __restrict__ e, float ce, int nt) {
  size_t i = (size_t)blockIdx.x * blockDim.x + threadIdx.x;
  if (i >= n) return;
  float v = ca * a[i];
  if (nt > 1) v += cb * b[i];
  if (nt > 2) v += cc * c[i];
  if (nt > 3) v += cd_ * d[i];
  if (nt > 4) v += ce * e[i];
  out[i] = v;
}

// kd = -clamp(div / trace_steps, -CLAMP, CLAMP)
__global__ void k_divfix(float* __restrict__ kd, const float* __restrict__ dv,
                         int n, float invTrace, float clampv) {
  int i = blockIdx.x * blockDim.x + threadIdx.x;
  if (i < n) {
    float v = dv[i] * invTrace;
    v = fminf(fmaxf(v, -clampv), clampv);
    kd[i] = -v;
  }
}

// rep = z ; lp[b] = -0.5*sum(z[b,:]^2) + ell[b]
__global__ void k_final(const float* __restrict__ z, const float* __restrict__ ell,
                        float* __restrict__ rep, float* __restrict__ lp, int Dd) {
  __shared__ float red[256];
  const int b = blockIdx.x;
  float s = 0.0f;
  for (int d = threadIdx.x; d < Dd; d += 256) {
    float v = z[(size_t)b * Dd + d];
    rep[(size_t)b * Dd + d] = v;
    s += v * v;
  }
  red[threadIdx.x] = s;
  __syncthreads();
  for (int off = 128; off > 0; off >>= 1) {
    if (threadIdx.x < off) red[threadIdx.x] += red[threadIdx.x + off];
    __syncthreads();
  }
  if (threadIdx.x == 0) lp[b] = -0.5f * red[0] + ell[b];
}

// ---------------------------------------------------------------------------
// Host orchestration
// ---------------------------------------------------------------------------
static inline int cdiv_i(size_t a, int b) { return (int)((a + b - 1) / b); }

extern "C" void kernel_launch(void* const* d_in, const int* in_sizes, int n_in,
                              void* d_out, int out_size, void* d_ws, size_t ws_size,
                              hipStream_t stream) {
  (void)n_in; (void)out_size; (void)ws_size;
  const float* x   = (const float*)d_in[0];
  const float* W1  = (const float*)d_in[1];
  const float* b1  = (const float*)d_in[2];
  const float* tw1 = (const float*)d_in[3];
  const float* W2  = (const float*)d_in[4];
  const float* b2  = (const float*)d_in[5];

  const int H  = in_sizes[2];               // 1024
  const int Dd = in_sizes[1] / H;           // 512
  const int Bn = in_sizes[0] / Dd;          // 2048
  const size_t BD = (size_t)Bn * Dd;
  const size_t BH = (size_t)Bn * H;
  const size_t sW = (size_t)Dd * H;

  // carve workspace (256B aligned chunks)
  uint8_t* base = (uint8_t*)d_ws;
  size_t off = 0;
  auto carve = [&](size_t bytes) -> void* {
    void* p = base + off;
    off = (off + bytes + 255) & ~(size_t)255;
    return p;
  };
  bf16* wsW1  = (bf16*)carve(sW * sizeof(bf16));   // [d][h]  (for g@W1^T)
  bf16* wsW1T = (bf16*)carve(sW * sizeof(bf16));   // [h][d]  (for z@W1)
  bf16* wsW2  = (bf16*)carve(sW * sizeof(bf16));   // [h][d]  (for e@W2^T)
  bf16* wsW2T = (bf16*)carve(sW * sizeof(bf16));   // [d][h]  (for h@W2)
  float* zst  = (float*)carve(BD * sizeof(float));
  float* ztmp = (float*)carve(BD * sizeof(float));
  float* hb   = (float*)carve(BH * sizeof(float));
  float* gb   = (float*)carve(BH * sizeof(float));
  float* eb   = (float*)carve(BD * sizeof(float));
  float* k1z  = (float*)carve(BD * sizeof(float));
  float* k2z  = (float*)carve(BD * sizeof(float));
  float* k3z  = (float*)carve(BD * sizeof(float));
  float* k4z  = (float*)carve(BD * sizeof(float));
  float* ell  = (float*)carve(Bn * sizeof(float));
  float* divb = (float*)carve(Bn * sizeof(float));
  float* kd1  = (float*)carve(Bn * sizeof(float));
  float* kd2  = (float*)carve(Bn * sizeof(float));
  float* kd3  = (float*)carve(Bn * sizeof(float));
  float* kd4  = (float*)carve(Bn * sizeof(float));

  const dim3 blk(256);

  // weight precompute (bf16 + transposed-layout copies)
  k_cvt <<<cdiv_i(sW, 256), blk, 0, stream>>>(W1, wsW1, sW);
  k_cvtT<<<cdiv_i(sW, 256), blk, 0, stream>>>(W1, wsW1T, Dd, H);
  k_cvt <<<cdiv_i(sW, 256), blk, 0, stream>>>(W2, wsW2, sW);
  k_cvtT<<<cdiv_i(sW, 256), blk, 0, stream>>>(W2, wsW2T, H, Dd);

  // state init: z = x, ell = 0
  k_comb<<<cdiv_i(BD, 256), blk, 0, stream>>>(zst, BD, x, 1.0f,
      nullptr, 0.f, nullptr, 0.f, nullptr, 0.f, nullptr, 0.f, 1);
  k_zero<<<cdiv_i(Bn, 256), blk, 0, stream>>>(ell, Bn);

  const dim3 gH(H / BN, Bn / BM);    // N = H GEMMs
  const dim3 gD(Dd / BN, Bn / BM);   // N = D GEMMs

  // augmented dynamics: (kz, kd) = f(t, zin)
  auto aug = [&](float t, float* zin, float* kz, float* kd, unsigned seed) {
    k_gemm<0><<<gH, blk, 0, stream>>>(zin, wsW1T, hb, Bn, H, Dd,
                                      b1, tw1, t, nullptr, nullptr, nullptr);
    k_gemm<1><<<gD, blk, 0, stream>>>(hb, wsW2T, kz, Bn, Dd, H,
                                      b2, nullptr, 0.f, nullptr, nullptr, nullptr);
    k_zero<<<cdiv_i(Bn, 256), blk, 0, stream>>>(divb, Bn);
    for (int i = 0; i < 2; ++i) {   // TRACE_STEPS
      k_gen_e<<<cdiv_i(BD, 256), blk, 0, stream>>>(eb, BD, seed * 2654435761u + (unsigned)i * 0x9E3779B9u + 7u);
      k_gemm<2><<<gH, blk, 0, stream>>>(eb, wsW2, gb, Bn, H, Dd,
                                        nullptr, nullptr, 0.f, hb, nullptr, nullptr);
      k_gemm<3><<<gD, blk, 0, stream>>>(gb, wsW1, nullptr, Bn, Dd, H,
                                        nullptr, nullptr, 0.f, nullptr, eb, divb);
    }
    k_divfix<<<cdiv_i(Bn, 256), blk, 0, stream>>>(kd, divb, Bn, 0.5f, 100.0f);
  };

  const int STEPS = 10;
  for (int s = 0; s < STEPS - 1; ++s) {
    const float t0 = (float)s / (float)(STEPS - 1);
    const float t1 = (float)(s + 1) / (float)(STEPS - 1);
    const float dt = t1 - t0;
    const unsigned sd = (unsigned)(s * 131 + 12345);

    // k1
    aug(t0, zst, k1z, kd1, sd + 0u);
    // z2 = z + dt*k1/3
    k_comb<<<cdiv_i(BD, 256), blk, 0, stream>>>(ztmp, BD, zst, 1.0f, k1z, dt / 3.0f,
        nullptr, 0.f, nullptr, 0.f, nullptr, 0.f, 2);
    aug(t0 + dt / 3.0f, ztmp, k2z, kd2, sd + 17u);
    // z3 = z + dt*k2 - dt*k1/3
    k_comb<<<cdiv_i(BD, 256), blk, 0, stream>>>(ztmp, BD, zst, 1.0f, k2z, dt, k1z, -dt / 3.0f,
        nullptr, 0.f, nullptr, 0.f, 3);
    aug(t0 + 2.0f * dt / 3.0f, ztmp, k3z, kd3, sd + 34u);
    // z4 = z + dt*(k1 - k2 + k3)
    k_comb<<<cdiv_i(BD, 256), blk, 0, stream>>>(ztmp, BD, zst, 1.0f, k1z, dt, k2z, -dt, k3z, dt,
        nullptr, 0.f, 4);
    aug(t1, ztmp, k4z, kd4, sd + 51u);
    // z += dt/8 * (k1 + 3k2 + 3k3 + k4)
    k_comb<<<cdiv_i(BD, 256), blk, 0, stream>>>(zst, BD, zst, 1.0f,
        k1z, dt * 0.125f, k2z, 3.0f * dt * 0.125f, k3z, 3.0f * dt * 0.125f, k4z, dt * 0.125f, 5);
    // ell += dt/8 * (kd1 + 3kd2 + 3kd3 + kd4)
    k_comb<<<cdiv_i(Bn, 256), blk, 0, stream>>>(ell, (size_t)Bn, ell, 1.0f,
        kd1, dt * 0.125f, kd2, 3.0f * dt * 0.125f, kd3, 3.0f * dt * 0.125f, kd4, dt * 0.125f, 5);
  }

  float* rep = (float*)d_out;
  float* lp  = rep + BD;
  k_final<<<Bn, blk, 0, stream>>>(zst, ell, rep, lp, Dd);
}
